// SpatialAttention_5884105196244
// MI455X (gfx1250) — matrix operand order repaired and checked by reading, not compiled, not run on hardware
//
#include <hip/hip_runtime.h>

typedef __attribute__((ext_vector_type(4)))  float  v4f;
typedef __attribute__((ext_vector_type(8)))  float  v8f;
typedef __attribute__((ext_vector_type(8)))  __bf16 v8bf;
typedef __attribute__((ext_vector_type(16))) __bf16 v16bf;

#define C_DIM 384
#define HW    4096
#define IMG_N 16

// ---------------------------------------------------------------------------
// helpers
// ---------------------------------------------------------------------------
static __device__ __forceinline__ v16bf cat8(v8bf lo, v8bf hi) {
  v16bf r;
#pragma unroll
  for (int i = 0; i < 8; ++i) { r[i] = lo[i]; r[i + 8] = hi[i]; }
  return r;
}

// ---------------------------------------------------------------------------
// Kernel 1: fused depthwise stack.
// One block per (n,c) 64x64 plane. x plane -> LDS, initial (5x5) -> LDS,
// spatial (sum of six 1-D large kernels) -> global workspace.
// Only x-read + spatial-write touch HBM (~200 MB total).
// ---------------------------------------------------------------------------
__global__ __launch_bounds__(256) void lka_depthwise(
    const float* __restrict__ x,
    const float* __restrict__ w0, const float* __restrict__ b0,
    const float* __restrict__ w1, const float* __restrict__ b1,
    const float* __restrict__ w2, const float* __restrict__ b2,
    const float* __restrict__ w3, const float* __restrict__ b3,
    const float* __restrict__ w4, const float* __restrict__ b4,
    const float* __restrict__ w5, const float* __restrict__ b5,
    const float* __restrict__ w6, const float* __restrict__ b6,
    float* __restrict__ sp)
{
  __shared__ float xs[4096];   // input plane
  __shared__ float ini[4096];  // initial = dw5x5(x)+b0

  const int tid = threadIdx.x;
  const int nc  = blockIdx.x;          // n*C + c
  const int c   = nc % C_DIM;
  const float* __restrict__ xp = x + (size_t)nc * HW;

  // cooperative plane load, b128 coalesced
#pragma unroll
  for (int i = 0; i < 4; ++i)
    *(v4f*)(xs + i * 1024 + tid * 4) = *(const v4f*)(xp + i * 1024 + tid * 4);
  __syncthreads();

  // --- 5x5 depthwise (pad 2) -> ini -------------------------------------
  float k5[25];
  {
    const float* p = w0 + c * 25;
#pragma unroll
    for (int i = 0; i < 25; ++i) k5[i] = p[i];
  }
  const float bias0 = b0[c];

#pragma unroll 1
  for (int i = 0; i < 16; ++i) {
    const int q  = tid + i * 256;
    const int yy = q >> 6, xx = q & 63;
    float s = bias0;
#pragma unroll
    for (int dy = -2; dy <= 2; ++dy) {
#pragma unroll
      for (int dx = -2; dx <= 2; ++dx) {
        const int iy = yy + dy, ix = xx + dx;
        const bool in = (iy >= 0) & (iy < 64) & (ix >= 0) & (ix < 64);
        s += k5[(dy + 2) * 5 + (dx + 2)] * (in ? xs[iy * 64 + ix] : 0.0f);
      }
    }
    ini[q] = s;
  }
  __syncthreads();

  // --- six 1-D depthwise kernels on ini, summed --------------------------
  float kh7[7], kv7[7], kh11[11], kv11[11], kh21[21], kv21[21];
  {
    const float* p1 = w1 + c * 7;
    const float* p2 = w2 + c * 7;
    const float* p3 = w3 + c * 11;
    const float* p4 = w4 + c * 11;
    const float* p5 = w5 + c * 21;
    const float* p6 = w6 + c * 21;
#pragma unroll
    for (int i = 0; i < 7;  ++i) { kh7[i]  = p1[i]; kv7[i]  = p2[i]; }
#pragma unroll
    for (int i = 0; i < 11; ++i) { kh11[i] = p3[i]; kv11[i] = p4[i]; }
#pragma unroll
    for (int i = 0; i < 21; ++i) { kh21[i] = p5[i]; kv21[i] = p6[i]; }
  }
  const float bsum = b1[c] + b2[c] + b3[c] + b4[c] + b5[c] + b6[c];
  float* __restrict__ spp = sp + (size_t)nc * HW;

#pragma unroll 1
  for (int i = 0; i < 16; ++i) {
    const int q  = tid + i * 256;
    const int yy = q >> 6, xx = q & 63;
    const float* row = ini + yy * 64;
    float s = bsum;
#pragma unroll
    for (int d = -3; d <= 3; ++d) {
      const int ix = xx + d;
      s += kh7[d + 3] * ((ix >= 0 && ix < 64) ? row[ix] : 0.0f);
    }
#pragma unroll
    for (int d = -3; d <= 3; ++d) {
      const int iy = yy + d;
      s += kv7[d + 3] * ((iy >= 0 && iy < 64) ? ini[iy * 64 + xx] : 0.0f);
    }
#pragma unroll
    for (int d = -5; d <= 5; ++d) {
      const int ix = xx + d;
      s += kh11[d + 5] * ((ix >= 0 && ix < 64) ? row[ix] : 0.0f);
    }
#pragma unroll
    for (int d = -5; d <= 5; ++d) {
      const int iy = yy + d;
      s += kv11[d + 5] * ((iy >= 0 && iy < 64) ? ini[iy * 64 + xx] : 0.0f);
    }
#pragma unroll
    for (int d = -10; d <= 10; ++d) {
      const int ix = xx + d;
      s += kh21[d + 10] * ((ix >= 0 && ix < 64) ? row[ix] : 0.0f);
    }
#pragma unroll
    for (int d = -10; d <= 10; ++d) {
      const int iy = yy + d;
      s += kv21[d + 10] * ((iy >= 0 && iy < 64) ? ini[iy * 64 + xx] : 0.0f);
    }
    spp[q] = s;
  }
}

// ---------------------------------------------------------------------------
// Kernel 2: 1x1 pointwise conv as WMMA GEMM.
// Out(384 x 65536) = WA(384 x 384) x SP(384 x 65536), + ba.
// bf16 hi/lo split precision: ab ~= ah*bh + ah*bl + al*bh (3 WMMAs / tile).
// Block tile 64(M) x 128(N), K-step 32; 8 waves in 2x4 grid, 2x2 16x16
// accumulators per wave.
// ---------------------------------------------------------------------------
#define BSTR 40  // padded LDS row stride for B (halfs); keeps 16B alignment

__global__ __launch_bounds__(256) void lka_pointwise_wmma(
    const float* __restrict__ sp,
    const float* __restrict__ wa,
    const float* __restrict__ ba,
    float* __restrict__ out)
{
  __shared__ __bf16 Ah[64 * 32];
  __shared__ __bf16 Al[64 * 32];
  __shared__ __bf16 Bh[128 * BSTR];
  __shared__ __bf16 Bl[128 * BSTR];

  const int tid  = threadIdx.x;
  const int wave = tid >> 5;
  const int lane = tid & 31;
  const int l16  = lane & 15;
  const int g    = lane >> 4;        // lane half
  const int wm   = wave >> 2;        // 0..1 (M)
  const int wn   = wave & 3;         // 0..3 (N)

  const int o0 = blockIdx.x * 64;            // output-channel tile
  const int pg = blockIdx.y * 128;           // flat pixel tile over N*HW
  const int n  = pg >> 12;                   // image
  const int p0 = pg & (HW - 1);              // pixel within image

  const float* __restrict__ spn  = sp  + (size_t)n * C_DIM * HW;
  float*       __restrict__ outn = out + (size_t)n * C_DIM * HW;

  v8f acc[2][2] = {};

  // cooperative-load index split
  const int arow = tid >> 2;          // 0..63  (A row = output channel)
  const int acol = (tid & 3) * 8;     // 0,8,16,24 (A col = k)
  const int brow = tid >> 3;          // 0..31  (B row = k)
  const int bcol = (tid & 7) * 16;    // 0..112 (B col = pixel)

  for (int k0 = 0; k0 < C_DIM; k0 += 32) {
    // ---- stage A tile (64x32 of WA) as bf16 hi/lo ----
    {
      const float* ap = wa + (size_t)(o0 + arow) * C_DIM + (k0 + acol);
      const v4f a0 = *(const v4f*)(ap);
      const v4f a1 = *(const v4f*)(ap + 4);
#pragma unroll
      for (int j = 0; j < 4; ++j) {
        const float v = a0[j];
        const __bf16 h = (__bf16)v;
        Ah[arow * 32 + acol + j] = h;
        Al[arow * 32 + acol + j] = (__bf16)(v - (float)h);
      }
#pragma unroll
      for (int j = 0; j < 4; ++j) {
        const float v = a1[j];
        const __bf16 h = (__bf16)v;
        Ah[arow * 32 + acol + 4 + j] = h;
        Al[arow * 32 + acol + 4 + j] = (__bf16)(v - (float)h);
      }
    }
    // ---- stage B tile (32x128 of SP), transposed to [pixel][k] ----
    {
      const float* bp = spn + (size_t)(k0 + brow) * HW + (p0 + bcol);
#pragma unroll
      for (int blk = 0; blk < 4; ++blk) {
        const v4f bv = *(const v4f*)(bp + blk * 4);
#pragma unroll
        for (int j = 0; j < 4; ++j) {
          const float v = bv[j];
          const __bf16 h = (__bf16)v;
          const int pr = bcol + blk * 4 + j;
          Bh[pr * BSTR + brow] = h;
          Bl[pr * BSTR + brow] = (__bf16)(v - (float)h);
        }
      }
    }
    __syncthreads();

    // ---- fragment assembly per documented CDNA5 16-bit layouts ----
    v16bf afh[2], afl[2], bfh[2], bfl[2];
#pragma unroll
    for (int mt = 0; mt < 2; ++mt) {
      const int row = wm * 32 + mt * 16 + l16;     // A row (M)
      const __bf16* base = &Ah[row * 32];
      // elements 0..7: K = 8g..8g+7 ; elements 8..15: K = 16+8g..23+8g
      afh[mt] = cat8(*(const v8bf*)(base + 8 * g),
                     *(const v8bf*)(base + 16 + 8 * g));
      const __bf16* basel = &Al[row * 32];
      afl[mt] = cat8(*(const v8bf*)(basel + 8 * g),
                     *(const v8bf*)(basel + 16 + 8 * g));
    }
#pragma unroll
    for (int nt = 0; nt < 2; ++nt) {
      const int prow = wn * 32 + nt * 16 + l16;    // B col (N), stored as row
      const __bf16* base = &Bh[prow * BSTR];  // K = 8g+i (i<8), 16+8g+(i-8)
      bfh[nt] = cat8(*(const v8bf*)(base + 8 * g), *(const v8bf*)(base + 16 + 8 * g));
      const __bf16* basel = &Bl[prow * BSTR];
      bfl[nt] = cat8(*(const v8bf*)(basel + 8 * g), *(const v8bf*)(basel + 16 + 8 * g));
    }

    // ---- 3-pass split-precision WMMA ----
#pragma unroll
    for (int mt = 0; mt < 2; ++mt) {
#pragma unroll
      for (int nt = 0; nt < 2; ++nt) {
        acc[mt][nt] = __builtin_amdgcn_wmma_f32_16x16x32_bf16(
            false, afh[mt], false, bfh[nt], (short)0, acc[mt][nt], false, false);
        acc[mt][nt] = __builtin_amdgcn_wmma_f32_16x16x32_bf16(
            false, afh[mt], false, bfl[nt], (short)0, acc[mt][nt], false, false);
        acc[mt][nt] = __builtin_amdgcn_wmma_f32_16x16x32_bf16(
            false, afl[mt], false, bfh[nt], (short)0, acc[mt][nt], false, false);
      }
    }
    __syncthreads();
  }

  // ---- epilogue: +bias, store (C/D layout: M = r + 8g, N = l16) ----
#pragma unroll
  for (int mt = 0; mt < 2; ++mt) {
#pragma unroll
    for (int r = 0; r < 8; ++r) {
      const int o = o0 + wm * 32 + mt * 16 + 8 * g + r;
      const float bias = ba[o];
      float* orow = outn + (size_t)o * HW;
#pragma unroll
      for (int nt = 0; nt < 2; ++nt) {
        const int p = p0 + wn * 32 + nt * 16 + l16;
        orow[p] = acc[mt][nt][r] + bias;
      }
    }
  }
}

// ---------------------------------------------------------------------------
// launch
// ---------------------------------------------------------------------------
extern "C" void kernel_launch(void* const* d_in, const int* in_sizes, int n_in,
                              void* d_out, int out_size, void* d_ws, size_t ws_size,
                              hipStream_t stream) {
  (void)in_sizes; (void)n_in; (void)out_size; (void)ws_size;
  const float* x  = (const float*)d_in[0];
  const float* w0 = (const float*)d_in[1];  const float* b0 = (const float*)d_in[2];
  const float* w1 = (const float*)d_in[3];  const float* b1 = (const float*)d_in[4];
  const float* w2 = (const float*)d_in[5];  const float* b2 = (const float*)d_in[6];
  const float* w3 = (const float*)d_in[7];  const float* b3 = (const float*)d_in[8];
  const float* w4 = (const float*)d_in[9];  const float* b4 = (const float*)d_in[10];
  const float* w5 = (const float*)d_in[11]; const float* b5 = (const float*)d_in[12];
  const float* w6 = (const float*)d_in[13]; const float* b6 = (const float*)d_in[14];
  const float* wa = (const float*)d_in[15]; const float* ba = (const float*)d_in[16];
  float* out = (float*)d_out;
  float* sp  = (float*)d_ws;   // spatial: 16*384*4096 f32 (~101 MB)

  lka_depthwise<<<dim3(IMG_N * C_DIM), dim3(256), 0, stream>>>(
      x, w0, b0, w1, b1, w2, b2, w3, b3, w4, b4, w5, b5, w6, b6, sp);

  lka_pointwise_wmma<<<dim3(C_DIM / 64, (IMG_N * HW) / 128), dim3(256), 0, stream>>>(
      sp, wa, ba, out);
}
